// MultiHeadAttention_12910671691777
// MI455X (gfx1250) — matrix operand order repaired and checked by reading, not compiled, not run on hardware
//
#include <hip/hip_runtime.h>
#include <hip/hip_bf16.h>

// MHA forward: B=32, T=448, C=1024, H=16, D=64
// f32->bf16 cvt -> WMMA QKV GEMM (async-LDS double buffered) -> flash-attn (WMMA)
//  -> WMMA proj GEMM (+bias, f32 out)
// Workspace: x_bf16(28MB)+w_attn_bf16(6MB)+w_proj_bf16(2MB)+qkv_bf16(84MB)+attn_bf16(28MB) ~148MB

typedef __attribute__((ext_vector_type(16))) __bf16 v16bf;
typedef __attribute__((ext_vector_type(8)))  float  v8f;

#define B_  32
#define T_  448
#define C_  1024
#define H_  16
#define D_  64
#define M_  (B_ * T_)     // 14336
#define N1_ (3 * C_)      // 3072

union FragU { v16bf v; uint4 q[2]; };

// A fragment (16x32 bf16, row-major source): lane%16 = row M, lane/16 selects K half.
__device__ inline v16bf load_a_frag(const __bf16* base, int ld, int lane) {
    int m = lane & 15, hi = lane >> 4;
    const __bf16* p = base + (size_t)m * ld + hi * 8;
    FragU f;
    f.q[0] = *reinterpret_cast<const uint4*>(p);
    f.q[1] = *reinterpret_cast<const uint4*>(p + 16);
    return f.v;
}

// B fragment (32x16 bf16) from [N,K] row-major: lane%16 = col N, lane/16 selects K half.
__device__ inline v16bf load_b_frag(const __bf16* base, int ld, int lane) {
    int n = lane & 15, hi = lane >> 4;
    const __bf16* p = base + (size_t)n * ld + hi * 8;
    FragU f;
    f.q[0] = *reinterpret_cast<const uint4*>(p);
    f.q[1] = *reinterpret_cast<const uint4*>(p + 16);
    return f.v;
}

__device__ inline v8f wmma_bf16(v16bf a, v16bf b, v8f c) {
    return __builtin_amdgcn_wmma_f32_16x16x32_bf16(false, a, false, b, (short)0, c, false, false);
}

__device__ inline float hmax16(float v) {
    #pragma unroll
    for (int m = 1; m < 16; m <<= 1) v = fmaxf(v, __shfl_xor(v, m, 32));
    return v;
}
__device__ inline float hsum16(float v) {
    #pragma unroll
    for (int m = 1; m < 16; m <<= 1) v += __shfl_xor(v, m, 32);
    return v;
}

// ASYNCcnt-tracked global->LDS 16B copy (GV mode: VDST = LDS offset, VADDR = 64b pair, SADDR=off)
__device__ inline void async_copy_b128(const void* gptr, void* lptr) {
    unsigned lds = (unsigned)(size_t)lptr;                 // low 32b of generic = LDS byte offset
    unsigned long long ga = (unsigned long long)(size_t)gptr;
    asm volatile("global_load_async_to_lds_b128 %0, %1, off" :: "v"(lds), "v"(ga) : "memory");
}

__global__ void f32_to_bf16_kernel(const float* __restrict__ in, __bf16* __restrict__ out, long n) {
    long i = (long)blockIdx.x * blockDim.x + threadIdx.x;
    long stride = (long)gridDim.x * blockDim.x;
    for (; i < n; i += stride) out[i] = (__bf16)in[i];
}

// GEMM: out[M,N] = A[M,K] @ W[N,K]^T (+bias). Block tile 256x128 (8 waves, 64x64/wave),
// K-step 32, double-buffered async global->LDS staging. M%256==0, N%128==0, K%32==0.
#define LDP 40   // padded LDS row length in bf16 (80B = 20 banks -> conflict-free frag reads)

template<bool OUT_BF16, bool BIAS>
__global__ __launch_bounds__(256) void gemm_bf16_wmma(
    const __bf16* __restrict__ A, const __bf16* __restrict__ W,
    void* __restrict__ outp, const float* __restrict__ bias,
    int M, int N, int K) {
    __shared__ __align__(16) __bf16 sA[2][256 * LDP];  // [m][k] rows of 32 + 8 pad
    __shared__ __align__(16) __bf16 sB[2][128 * LDP];  // [n][k] rows of 32 + 8 pad

    int tid = threadIdx.x, wave = tid >> 5, lane = tid & 31;
    int wm = wave >> 1, wn = wave & 1;
    int m_blk = blockIdx.y * 256, n_blk = blockIdx.x * 128;
    int m = lane & 15, hi = lane >> 4;

    // stage one 256x32 A tile + 128x32 W tile: 6 x b128 async ops per thread (ASYNCcnt += 6)
    auto issue_stage = [&](int buf, int k0) {
        #pragma unroll
        for (int j = 0; j < 4; ++j) {                   // A: 1024 chunks of 16B
            int c = tid + 256 * j;
            int row = c >> 2, cc = c & 3;
            async_copy_b128(A + (size_t)(m_blk + row) * K + k0 + cc * 8,
                            &sA[buf][row * LDP + cc * 8]);
        }
        #pragma unroll
        for (int j = 0; j < 2; ++j) {                   // B: 512 chunks of 16B
            int c = tid + 256 * j;
            int row = c >> 2, cc = c & 3;
            async_copy_b128(W + (size_t)(n_blk + row) * K + k0 + cc * 8,
                            &sB[buf][row * LDP + cc * 8]);
        }
    };

    v8f zero = {};
    v8f acc[4][4];
    #pragma unroll
    for (int i = 0; i < 4; ++i)
        #pragma unroll
        for (int j = 0; j < 4; ++j) acc[i][j] = zero;

    issue_stage(0, 0);
    int nk = K / 32;
    for (int i = 0; i < nk; ++i) {
        int buf = i & 1;
        if (i + 1 < nk) {
            issue_stage(buf ^ 1, (i + 1) * 32);
            asm volatile("s_wait_asynccnt 0x6" ::: "memory");  // stage i done, i+1 in flight
        } else {
            asm volatile("s_wait_asynccnt 0x0" ::: "memory");
        }
        __syncthreads();                                        // all waves' stage-i data visible

        v16bf af[4], bfr[4];
        #pragma unroll
        for (int mt = 0; mt < 4; ++mt) {
            const __bf16* p = &sA[buf][(wm * 64 + mt * 16 + m) * LDP + hi * 8];
            FragU f;
            f.q[0] = *reinterpret_cast<const uint4*>(p);
            f.q[1] = *reinterpret_cast<const uint4*>(p + 16);
            af[mt] = f.v;
        }
        #pragma unroll
        for (int nt = 0; nt < 4; ++nt) {
            const __bf16* p = &sB[buf][(wn * 64 + nt * 16 + m) * LDP + hi * 8];
            FragU f;
            f.q[0] = *reinterpret_cast<const uint4*>(p);
            f.q[1] = *reinterpret_cast<const uint4*>(p + 16);
            bfr[nt] = f.v;
        }
        #pragma unroll
        for (int mt = 0; mt < 4; ++mt)
            #pragma unroll
            for (int nt = 0; nt < 4; ++nt)
                acc[mt][nt] = wmma_bf16(af[mt], bfr[nt], acc[mt][nt]);

        __syncthreads();                                        // before buf is overwritten
    }

    #pragma unroll
    for (int mt = 0; mt < 4; ++mt) {
        #pragma unroll
        for (int nt = 0; nt < 4; ++nt) {
            int row = m_blk + wm * 64 + 16 * mt + 8 * hi;
            int col = n_blk + wn * 64 + 16 * nt + m;
            float bv = BIAS ? bias[col] : 0.0f;
            #pragma unroll
            for (int r = 0; r < 8; ++r) {
                float v = acc[mt][nt][r] + bv;
                if (OUT_BF16)
                    ((__bf16*)outp)[(size_t)(row + r) * N + col] = (__bf16)v;
                else
                    ((float*)outp)[(size_t)(row + r) * N + col] = v;
            }
        }
    }
}

// Flash attention. qkv: [B*T, 3C] bf16 rows = [q|k|v]. One wave per 16-row query tile
// per (b,h); per-wave private LDS (divergent causal trip counts -> no cross-wave barriers).
__global__ __launch_bounds__(128) void attn_wmma(
    const __bf16* __restrict__ qkv, const unsigned char* __restrict__ pmask,
    __bf16* __restrict__ attn) {
    __shared__ __align__(16) __bf16 sS[4][16 * 32];  // per-wave P tile (D-frag -> A-frag relayout)
    __shared__ __align__(16) __bf16 sV[4][32 * 64];  // per-wave staged V tile

    int tid = threadIdx.x, wave = tid >> 5, lane = tid & 31;
    int n = lane & 15, hi = lane >> 4;
    int qi = blockIdx.x * 4 + wave;     // query tile 0..27
    int h  = blockIdx.y;
    int b  = blockIdx.z;

    const __bf16* qp = qkv + (size_t)(b * T_ + qi * 16) * N1_ + h * D_;
    v16bf aq0 = load_a_frag(qp,      N1_, lane);   // d in [0,32)
    v16bf aq1 = load_a_frag(qp + 32, N1_, lane);   // d in [32,64)

    float mrow[8], lrow[8];
    v8f zero = {};
    v8f acc[4];
    #pragma unroll
    for (int r = 0; r < 8; ++r) { mrow[r] = -1e30f; lrow[r] = 0.0f; }
    #pragma unroll
    for (int dt = 0; dt < 4; ++dt) acc[dt] = zero;

    const float scale = 0.03125f;       // C^-0.5 = 1024^-0.5
    int trow = qi * 16 + 8 * hi;
    int nblk = (qi >> 1) + 1;

    for (int blk = 0; blk < nblk; ++blk) {
        int s0 = blk * 32;
        const __bf16* kp0 = qkv + (size_t)(b * T_ + s0) * N1_ + C_ + h * D_;
        const __bf16* kp1 = kp0 + (size_t)16 * N1_;

        v8f sc0 = wmma_bf16(aq1, load_b_frag(kp0 + 32, N1_, lane),
                  wmma_bf16(aq0, load_b_frag(kp0,      N1_, lane), zero));
        v8f sc1 = wmma_bf16(aq1, load_b_frag(kp1 + 32, N1_, lane),
                  wmma_bf16(aq0, load_b_frag(kp1,      N1_, lane), zero));

        {   // stage V[s0..s0+31][0..63] (1 row/lane, 128B)
            const uint4* src = reinterpret_cast<const uint4*>(
                qkv + (size_t)(b * T_ + s0 + lane) * N1_ + 2 * C_ + h * D_);
            uint4* dst = reinterpret_cast<uint4*>(&sV[wave][lane * 64]);
            #pragma unroll
            for (int j = 0; j < 8; ++j) dst[j] = src[j];
        }

        #pragma unroll
        for (int r = 0; r < 8; ++r) {
            int t  = trow + r;
            int c0 = s0 + n, c1 = s0 + 16 + n;
            float v0 = ((c0 > t) || pmask[b * T_ + c0]) ? -1e30f : sc0[r] * scale;
            float v1 = ((c1 > t) || pmask[b * T_ + c1]) ? -1e30f : sc1[r] * scale;
            float bm   = hmax16(fmaxf(v0, v1));
            float mnew = fmaxf(mrow[r], bm);
            float scl  = __expf(mrow[r] - mnew);
            float e0   = __expf(v0 - mnew);
            float e1   = __expf(v1 - mnew);
            lrow[r] = lrow[r] * scl + hsum16(e0 + e1);
            mrow[r] = mnew;
            #pragma unroll
            for (int dt = 0; dt < 4; ++dt) acc[dt][r] *= scl;
            sS[wave][(8 * hi + r) * 32 + n]      = (__bf16)e0;
            sS[wave][(8 * hi + r) * 32 + 16 + n] = (__bf16)e1;
        }

        v16bf pf;
        {   // reload P in A-fragment layout (same-wave LDS ops are in-order)
            const __bf16* srow = &sS[wave][(lane & 15) * 32 + hi * 8];
            FragU f;
            f.q[0] = *reinterpret_cast<const uint4*>(srow);
            f.q[1] = *reinterpret_cast<const uint4*>(srow + 16);
            pf = f.v;
        }

        #pragma unroll
        for (int dt = 0; dt < 4; ++dt) {
            v16bf vf;
            #pragma unroll
            for (int j = 0; j < 16; ++j)
                vf[j] = sV[wave][(hi * 8 + (j & 7) + (j >> 3) * 16) * 64 + dt * 16 + n];
            acc[dt] = wmma_bf16(pf, vf, acc[dt]);
        }
    }

    #pragma unroll
    for (int dt = 0; dt < 4; ++dt) {
        #pragma unroll
        for (int r = 0; r < 8; ++r) {
            float o = acc[dt][r] / lrow[r];
            attn[(size_t)(b * T_ + trow + r) * C_ + h * D_ + dt * 16 + n] = (__bf16)o;
        }
    }
}

extern "C" void kernel_launch(void* const* d_in, const int* in_sizes, int n_in,
                              void* d_out, int out_size, void* d_ws, size_t ws_size,
                              hipStream_t stream) {
    (void)in_sizes; (void)n_in; (void)out_size; (void)ws_size;
    const float*         x      = (const float*)d_in[0];
    const unsigned char* pmask  = (const unsigned char*)d_in[1]; // bool mask
    const float*         w_attn = (const float*)d_in[2];
    const float*         w_proj = (const float*)d_in[3];
    const float*         b_proj = (const float*)d_in[4];

    char* ws = (char*)d_ws;
    __bf16* x_bf   = (__bf16*)ws; ws += (size_t)M_  * C_  * 2;
    __bf16* wa_bf  = (__bf16*)ws; ws += (size_t)N1_ * C_  * 2;
    __bf16* wp_bf  = (__bf16*)ws; ws += (size_t)C_  * C_  * 2;
    __bf16* qkv_bf = (__bf16*)ws; ws += (size_t)M_  * N1_ * 2;
    __bf16* at_bf  = (__bf16*)ws;

    f32_to_bf16_kernel<<<2048, 256, 0, stream>>>(x,      x_bf,  (long)M_  * C_);
    f32_to_bf16_kernel<<<1024, 256, 0, stream>>>(w_attn, wa_bf, (long)N1_ * C_);
    f32_to_bf16_kernel<<<512,  256, 0, stream>>>(w_proj, wp_bf, (long)C_  * C_);

    // qkv = x @ w_attn^T -> bf16
    gemm_bf16_wmma<true, false><<<dim3(N1_ / 128, M_ / 256), 256, 0, stream>>>(
        x_bf, wa_bf, qkv_bf, nullptr, M_, N1_, C_);

    // flash attention -> bf16 [B*T, C]
    attn_wmma<<<dim3(7, H_, B_), 128, 0, stream>>>(qkv_bf, pmask, at_bf);

    // out = attn @ w_proj^T + b_proj -> f32
    gemm_bf16_wmma<false, true><<<dim3(C_ / 128, M_ / 256), 256, 0, stream>>>(
        at_bf, wp_bf, d_out, b_proj, M_, C_, C_);
}
